// ThreeBodyConv_53334903882518
// MI455X (gfx1250) — matrix operand order repaired and checked by reading, not compiled, not run on hardware
//
#include <hip/hip_runtime.h>

// ---------------- problem constants (from reference) ----------------
#define N_ATOMS 50000
#define F_DIM   64
#define NB      16
#define E_EDGES 1600000
#define T_TRI   1000000
#define IN_DIM  208          // F + 2*FE + NB
#define K_PAD   224          // pad K to 7 * 32 for bf16 WMMA
#define N_OUT   128          // 2*F
#define KSTEPS  7
#define NTILES  8            // 128 / 16
#define WAVES   8            // 256-thread block = 8 wave32
#define BN_EPS  1e-5f
#define INV_VAR2 (1.0f / (0.15f * 0.15f))

#define BP_ELEMS (KSTEPS * NTILES * 32 * 16)   // 28672 bf16 swizzled weights
#define ATILE_BYTES (WAVES * 16 * K_PAD * 2)   // 57344
#define BTILE_BYTES (BP_ELEMS * 2)             // 57344
#define DYN_LDS_BYTES (ATILE_BYTES + BTILE_BYTES)

typedef __bf16 bf16;
typedef bf16  v8bf  __attribute__((ext_vector_type(8)));
typedef bf16  v16bf __attribute__((ext_vector_type(16)));
typedef float v8f   __attribute__((ext_vector_type(8)));
typedef unsigned u32x4 __attribute__((ext_vector_type(4)));
typedef int      i32x4 __attribute__((ext_vector_type(4)));
typedef int      i32x8 __attribute__((ext_vector_type(8)));

// ---------------- workspace byte offsets ----------------
#define WS_BN1_STATS 0        // 256 floats: sum[128], sumsq[128]
#define WS_BN1_AFF   1024     // 256 floats: scale[128], shift[128]
#define WS_BN2_STATS 2048     // 128 floats: sum[64], sumsq[64]
#define WS_BN2_AFF   2560     // 128 floats: scale[64], shift[64]
#define WS_BPRIME    3072     // 28672 bf16 (57344 B): WMMA-B-fragment swizzled weights
#define WS_AGGR      60416    // N_ATOMS * F_DIM floats (segment-sum accumulator)

__device__ __forceinline__ bf16 f2bf(float f) {
  // round-to-nearest-even fp32 -> bf16
  unsigned u = __builtin_bit_cast(unsigned, f);
  unsigned r = u + 0x7fffu + ((u >> 16) & 1u);
  unsigned short h = (unsigned short)(r >> 16);
  return __builtin_bit_cast(bf16, h);
}

__device__ __forceinline__ float softplus_f(float x) {
  return (x > 20.f) ? x : __logf(1.f + __expf(x));
}

// Issue a TDM DMA of the 57344-byte swizzled weight block: global -> LDS.
// D# per CDNA5 ISA: group0 {count=1, lds_addr, global_addr,
// type=2}, group1 {data_size=2B, tensor_dim0=tile_dim0=28672, tile_dim1=1,
// dim0_stride=28672}; groups 2/3 zero (<=2D tensor). Tracked by TENSORcnt.
__device__ __forceinline__ void tdm_load_btile(const bf16* gsrc, bf16* lds_dst) {
  unsigned long long gaddr = (unsigned long long)(size_t)gsrc;
  unsigned lds_off = (unsigned)(size_t)lds_dst;   // flat LDS addr low 32 bits
  u32x4 g0;
  g0[0] = 1u;                                        // count=1
  g0[1] = lds_off;                                   // lds_addr
  g0[2] = (unsigned)(gaddr & 0xffffffffu);           // global_addr[31:0]
  g0[3] = (unsigned)((gaddr >> 32) & 0x01ffffffu) | (2u << 30);  // [56:32] | type=2
  i32x8 g1;
  g1[0] = (int)(1u << 16);                           // data_size=1 (2 bytes)
  g1[1] = (int)((BP_ELEMS & 0xffffu) << 16);         // tensor_dim0[15:0]
  g1[2] = (int)(((BP_ELEMS >> 16) & 0xffffu) | (1u << 16));  // dim0[31:16], dim1=1
  g1[3] = (int)((BP_ELEMS & 0xffffu) << 16);         // tile_dim0 = 28672
  g1[4] = 1;                                         // tile_dim1 = 1
  g1[5] = (int)BP_ELEMS;                             // tensor_dim0_stride
  g1[6] = 0;
  g1[7] = 0;
  i32x4 gz = {0, 0, 0, 0};
#if __clang_major__ >= 23
  i32x8 gz8 = {0, 0, 0, 0, 0, 0, 0, 0};
  __builtin_amdgcn_tensor_load_to_lds(g0, g1, gz, gz, gz8, 0);
#else
  __builtin_amdgcn_tensor_load_to_lds(g0, g1, gz, gz, 0);
#endif
}

// Pre-swizzle fc_w (128 x 208 fp32, out-major) into bf16 B-fragments.
// Fragment (kstep, nt): 32 lanes x 16 bf16; lane n<16 holds column nt*16+n,
// K = kstep*32 + 0..7,16..23 ; lane n>=16 holds same column, K = kstep*32 + 8..15,24..31.
__global__ void prep_b_kernel(const float* __restrict__ fc_w, bf16* __restrict__ bp) {
  int kstep = blockIdx.x;                 // 0..6
  int nt    = threadIdx.x >> 5;           // 0..7
  int lane  = threadIdx.x & 31;
  int col   = nt * 16 + (lane & 15);
  int kbase = kstep * 32 + ((lane >> 4) * 8);
  bf16* dst = bp + (((kstep * NTILES + nt) * 32) + lane) * 16;
#pragma unroll
  for (int j = 0; j < 16; ++j) {
    int k = kbase + j + ((j >> 3) << 3);
    float v = (k < IN_DIM) ? fc_w[col * IN_DIM + k] : 0.0f;
    dst[j] = f2bf(v);
  }
}

// Fused gather + bf16 WMMA GEMM. PASS 1: accumulate BN1 column stats only.
// PASS 2: apply BN1 affine, gate*softplus, cutoff scaling, atomic segment-sum.
template <int PASS>
__global__ __launch_bounds__(256)
void triplet_gemm_kernel(const float* __restrict__ atom_fea,
                         const float* __restrict__ edge_fea,
                         const float* __restrict__ r_ij,
                         const float* __restrict__ dist,
                         const float* __restrict__ cutoff_w,
                         const float* __restrict__ fc_b,
                         const int*   __restrict__ edge_index,
                         const int*   __restrict__ triplet_idx,
                         const bf16*  __restrict__ bprime,
                         float*       __restrict__ bn1_stats,
                         const float* __restrict__ bn1_aff,
                         float*       __restrict__ aggr) {
  extern __shared__ char smem[];
  bf16* Atile = (bf16*)smem;                         // [WAVES*16][K_PAD]
  bf16* Btile = (bf16*)(smem + ATILE_BYTES);         // staged weight fragments

  __shared__ int   s_ctr[WAVES][16];
  __shared__ int   s_e1[WAVES][16];
  __shared__ int   s_e2[WAVES][16];
  __shared__ float s_cw[WAVES][16];
  __shared__ float s_val[WAVES][16];
  __shared__ float s_sum[N_OUT];
  __shared__ float s_sq[N_OUT];

  const int tid  = threadIdx.x;
  const int w    = tid >> 5;       // wave in block
  const int lane = tid & 31;
  const int tile = blockIdx.x * WAVES + w;   // 16-triplet tile id

  // Wave 0 kicks off the TDM weight DMA (EXEC-independent, async on TENSORcnt)
  // while all waves run the gather/convert phases.
  if (w == 0) {
    tdm_load_btile(bprime, Btile);
  }

  if (PASS == 1 && tid < N_OUT) { s_sum[tid] = 0.f; s_sq[tid] = 0.f; }

  // ---- phase 1: triplet meta + angular basis (lanes 0..15 each own a row) ----
  if (lane < 16) {
    const int m = lane;
    const long long t = (long long)tile * 16 + m;
    const bool valid = (t < (long long)T_TRI);
    int e1 = 0, e2 = 0;
    if (valid) { e1 = triplet_idx[t]; e2 = triplet_idx[(long long)T_TRI + t]; }
    const float x1 = r_ij[e1 * 3 + 0], y1 = r_ij[e1 * 3 + 1], z1 = r_ij[e1 * 3 + 2];
    const float x2 = r_ij[e2 * 3 + 0], y2 = r_ij[e2 * 3 + 1], z2 = r_ij[e2 * 3 + 2];
    const float d1 = fmaxf(dist[e1], 1e-8f), d2 = fmaxf(dist[e2], 1e-8f);
    float cosang = (x1 * x2 + y1 * y2 + z1 * z2) / (d1 * d2);
    cosang = fminf(1.f, fmaxf(-1.f, cosang));
    const float cw  = cutoff_w[e1] * cutoff_w[e2] * (valid ? 1.f : 0.f);
    const int   ctr = edge_index[E_EDGES + e1];   // dst[e1]
    s_ctr[w][m] = ctr; s_e1[w][m] = e1; s_e2[w][m] = e2;
    s_cw[w][m]  = cw;  s_val[w][m] = valid ? 1.f : 0.f;
    // warm the gather rows (global_prefetch_b8)
    __builtin_prefetch(&atom_fea[(long long)ctr * F_DIM], 0, 3);
    __builtin_prefetch(&edge_fea[(long long)e1 * F_DIM], 0, 3);
    __builtin_prefetch(&edge_fea[(long long)e2 * F_DIM], 0, 3);
    bf16* arow = Atile + (size_t)(w * 16 + m) * K_PAD;
#pragma unroll
    for (int j = 0; j < NB; ++j) {
      const float c = -1.f + (2.f / 15.f) * (float)j;
      const float d = cosang - c;
      arow[3 * F_DIM + j] = f2bf(__expf(-d * d * INV_VAR2));
    }
#pragma unroll
    for (int j = IN_DIM; j < K_PAD; ++j) arow[j] = f2bf(0.f);
  }
  __syncthreads();

  // ---- phase 2: gather fp32 rows -> bf16 A tile (2 lanes per row) ----
  {
    const int r = lane >> 1, half = lane & 1;
    bf16* arow = Atile + (size_t)(w * 16 + r) * K_PAD;
#pragma unroll
    for (int seg = 0; seg < 3; ++seg) {
      const int idx = (seg == 0) ? s_ctr[w][r] : (seg == 1) ? s_e1[w][r] : s_e2[w][r];
      const float* base = (seg == 0) ? atom_fea : edge_fea;
      const float4* src =
          reinterpret_cast<const float4*>(base + (long long)idx * F_DIM) + half * 8;
      bf16* dstrow = arow + seg * F_DIM + half * 32;
#pragma unroll
      for (int q = 0; q < 4; ++q) {
        const float4 a = src[2 * q], b = src[2 * q + 1];
        v8bf v;
        v[0] = f2bf(a.x); v[1] = f2bf(a.y); v[2] = f2bf(a.z); v[3] = f2bf(a.w);
        v[4] = f2bf(b.x); v[5] = f2bf(b.y); v[6] = f2bf(b.z); v[7] = f2bf(b.w);
        *reinterpret_cast<v8bf*>(dstrow + q * 8) = v;
      }
    }
  }
  // Wave 0 waits for the TDM DMA so the barrier release implies B is in LDS.
  if (w == 0) {
    __builtin_amdgcn_s_wait_tensorcnt(0);
  }
  __syncthreads();

  // ---- phase 3: 16x128 = 8 x WMMA(16x16), K = 7 steps of 32 (bf16) ----
  v8f acc[NTILES] = {};
  const int r16 = lane & 15;
  const int hi  = lane >> 4;
  const bf16* arow = Atile + (size_t)(w * 16 + r16) * K_PAD;
#pragma unroll
  for (int ks = 0; ks < KSTEPS; ++ks) {
    // A-fragment lane layout (ISA 7.12.2): lanes 0-15 K 0..7 & 16..23, lanes 16-31 K 8..15 & 24..31
    const int kb = ks * 32 + hi * 8;
    const v8bf lo = *reinterpret_cast<const v8bf*>(arow + kb);
    const v8bf hv = *reinterpret_cast<const v8bf*>(arow + kb + 16);
    const v16bf afrag = __builtin_shufflevector(lo, hv, 0, 1, 2, 3, 4, 5, 6, 7,
                                                8, 9, 10, 11, 12, 13, 14, 15);
#pragma unroll
    for (int nt = 0; nt < NTILES; ++nt) {
      const v16bf bfrag = *reinterpret_cast<const v16bf*>(
          Btile + (((ks * NTILES + nt) * 32) + lane) * 16);
      acc[nt] = __builtin_amdgcn_wmma_f32_16x16x32_bf16(
          false, afrag, false, bfrag, (short)0, acc[nt], false, false);
    }
  }

  // ---- epilogue: C/D layout: VGPR v, lane n<16 -> (M=v, N=n); n>=16 -> (M=v+8, N=n-16)
  float mrow[8], cwrow[8];
  int crow[8];
#pragma unroll
  for (int v = 0; v < 8; ++v) {
    const int m = v + 8 * hi;
    mrow[v]  = s_val[w][m];
    cwrow[v] = s_cw[w][m];
    crow[v]  = s_ctr[w][m];
  }

  if (PASS == 1) {
    // BN1 batch statistics of y = x*W^T + b (masked for tail rows)
#pragma unroll
    for (int nt = 0; nt < NTILES; ++nt) {
      const int col = nt * 16 + r16;
      const float bias = fc_b[col];
      float s = 0.f, q = 0.f;
#pragma unroll
      for (int v = 0; v < 8; ++v) {
        const float y = (acc[nt][v] + bias) * mrow[v];
        s += y; q += y * y;
      }
      atomicAdd(&s_sum[col], s);   // ds_add_f32
      atomicAdd(&s_sq[col], q);
    }
    __syncthreads();
    if (tid < N_OUT) {
      atomicAdd(&bn1_stats[tid], s_sum[tid]);
      atomicAdd(&bn1_stats[N_OUT + tid], s_sq[tid]);
    }
  } else {
    // BN1 affine + sigmoid(gate)*softplus(core)*cutoff -> atomic segment sum
    const float* scale = bn1_aff;
    const float* shift = bn1_aff + N_OUT;
#pragma unroll
    for (int nt = 0; nt < 4; ++nt) {
      const int cg = nt * 16 + r16;   // gate column (0..63)
      const int cc = cg + 64;         // core column
      const float bg = fc_b[cg], bc = fc_b[cc];
      const float sg = scale[cg], hg = shift[cg];
      const float sc = scale[cc], hc = shift[cc];
#pragma unroll
      for (int v = 0; v < 8; ++v) {
        const float yg = (acc[nt][v] + bg) * sg + hg;
        const float yc = (acc[nt + 4][v] + bc) * sc + hc;
        const float gate = 1.f / (1.f + __expf(-yg));
        const float msg  = gate * softplus_f(yc) * cwrow[v];  // cw==0 masks tail
        atomicAdd(&aggr[(long long)crow[v] * F_DIM + cg], msg);  // global_atomic_add_f32
      }
    }
  }
}

__global__ void bn1_finalize_kernel(const float* __restrict__ stats,
                                    const float* __restrict__ gamma,
                                    const float* __restrict__ beta,
                                    float* __restrict__ aff) {
  const int c = threadIdx.x;
  if (c < N_OUT) {
    const float mean = stats[c] / (float)T_TRI;
    const float var  = stats[N_OUT + c] / (float)T_TRI - mean * mean;
    const float sc   = gamma[c] * rsqrtf(var + BN_EPS);
    aff[c]          = sc;
    aff[N_OUT + c]  = beta[c] - mean * sc;
  }
}

__global__ void bn2_stats_kernel(const float* __restrict__ aggr,
                                 float* __restrict__ stats) {
  __shared__ float ssum[F_DIM], ssq[F_DIM];
  if (threadIdx.x < F_DIM) { ssum[threadIdx.x] = 0.f; ssq[threadIdx.x] = 0.f; }
  __syncthreads();
  const int gid = blockIdx.x * blockDim.x + threadIdx.x;
  const int col = gid & (F_DIM - 1);
  int row = gid >> 6;
  const int rstride = (gridDim.x * blockDim.x) >> 6;
  float s = 0.f, q = 0.f;
  for (; row < N_ATOMS; row += rstride) {
    const float v = aggr[(long long)row * F_DIM + col];
    s += v; q += v * v;
  }
  atomicAdd(&ssum[col], s);
  atomicAdd(&ssq[col], q);
  __syncthreads();
  if (threadIdx.x < F_DIM) {
    atomicAdd(&stats[threadIdx.x], ssum[threadIdx.x]);
    atomicAdd(&stats[F_DIM + threadIdx.x], ssq[threadIdx.x]);
  }
}

__global__ void bn2_finalize_kernel(const float* __restrict__ stats,
                                    const float* __restrict__ gamma,
                                    const float* __restrict__ beta,
                                    float* __restrict__ aff) {
  const int c = threadIdx.x;
  if (c < F_DIM) {
    const float mean = stats[c] / (float)N_ATOMS;
    const float var  = stats[F_DIM + c] / (float)N_ATOMS - mean * mean;
    const float sc   = gamma[c] * rsqrtf(var + BN_EPS);
    aff[c]          = sc;
    aff[F_DIM + c]  = beta[c] - mean * sc;
  }
}

__global__ void output_kernel(const float* __restrict__ atom_fea,
                              const float* __restrict__ aggr,
                              const float* __restrict__ aff,
                              float* __restrict__ out) {
  const long long i = (long long)blockIdx.x * blockDim.x + threadIdx.x;
  if (i >= (long long)N_ATOMS * F_DIM) return;
  const int c = (int)(i & (F_DIM - 1));
  const float y = aggr[i] * aff[c] + aff[F_DIM + c];
  out[i] = softplus_f(atom_fea[i] + y);
}

extern "C" void kernel_launch(void* const* d_in, const int* in_sizes, int n_in,
                              void* d_out, int out_size, void* d_ws, size_t ws_size,
                              hipStream_t stream) {
  const float* atom_fea   = (const float*)d_in[0];
  const float* edge_fea   = (const float*)d_in[1];
  const float* r_ij       = (const float*)d_in[2];
  const float* dist       = (const float*)d_in[3];
  const float* cutoff_w   = (const float*)d_in[4];
  const float* fc_w       = (const float*)d_in[5];
  const float* fc_b       = (const float*)d_in[6];
  const float* bn1_gamma  = (const float*)d_in[7];
  const float* bn1_beta   = (const float*)d_in[8];
  const float* bn2_gamma  = (const float*)d_in[9];
  const float* bn2_beta   = (const float*)d_in[10];
  const int*   edge_index = (const int*)d_in[11];
  const int*   triplet_idx= (const int*)d_in[12];
  float*       out        = (float*)d_out;

  char* ws = (char*)d_ws;
  float* bn1_stats = (float*)(ws + WS_BN1_STATS);
  float* bn1_aff   = (float*)(ws + WS_BN1_AFF);
  float* bn2_stats = (float*)(ws + WS_BN2_STATS);
  float* bn2_aff   = (float*)(ws + WS_BN2_AFF);
  bf16*  bprime    = (bf16*)(ws + WS_BPRIME);
  float* aggr      = (float*)(ws + WS_AGGR);

  // deterministic zero-init of stats + segment-sum accumulator (capture-safe)
  hipMemsetAsync(ws, 0, WS_BPRIME, stream);
  hipMemsetAsync(aggr, 0, (size_t)N_ATOMS * F_DIM * sizeof(float), stream);

  prep_b_kernel<<<KSTEPS, 256, 0, stream>>>(fc_w, bprime);

  const int gemm_grid = (T_TRI + 16 * WAVES - 1) / (16 * WAVES);  // 7813
  triplet_gemm_kernel<1><<<gemm_grid, 256, DYN_LDS_BYTES, stream>>>(
      atom_fea, edge_fea, r_ij, dist, cutoff_w, fc_b, edge_index, triplet_idx,
      bprime, bn1_stats, bn1_aff, aggr);
  bn1_finalize_kernel<<<1, 128, 0, stream>>>(bn1_stats, bn1_gamma, bn1_beta, bn1_aff);
  triplet_gemm_kernel<2><<<gemm_grid, 256, DYN_LDS_BYTES, stream>>>(
      atom_fea, edge_fea, r_ij, dist, cutoff_w, fc_b, edge_index, triplet_idx,
      bprime, bn1_stats, bn1_aff, aggr);

  bn2_stats_kernel<<<256, 256, 0, stream>>>(aggr, bn2_stats);
  bn2_finalize_kernel<<<1, 64, 0, stream>>>(bn2_stats, bn2_gamma, bn2_beta, bn2_aff);

  const long long total = (long long)N_ATOMS * F_DIM;
  output_kernel<<<(int)((total + 255) / 256), 256, 0, stream>>>(atom_fea, aggr, bn2_aff, out);
}
